// GCN_RES_25134148616264
// MI455X (gfx1250) — matrix operand order repaired and checked by reading, not compiled, not run on hardware
//
#include <hip/hip_runtime.h>

// ---------------------------------------------------------------------------
// GCN w/ residual on MI455X (gfx1250, wave32).
// Pipeline (all on `stream`, deterministic, graph-capturable):
//   1. zero agg1[N]
//   2. scatter1: agg1[dst] += feat[src]*mask1[src]            (f32 atomics)
//   3. h1 = relu(agg1 * W1 + b1)                              [N,64]
//   4. zero agg[N*64]; scatter2: agg[dst,:] += h1[src,:]      (f32 atomics)
//   5. WMMA GEMM2 fused: x = relu(h1 + agg@W2 + b2) * mask2   [N,64]
//   6. zero agg[N*64]; scatter3: agg[dst,:] += x[src,:]
//   7. WMMA GEMM3: out = agg@W3 + b3                          [N,128]
// ---------------------------------------------------------------------------

typedef _Float16 v16h __attribute__((ext_vector_type(16)));
typedef float    v8f  __attribute__((ext_vector_type(8)));

__device__ __forceinline__ void atomic_add_f32(float* p, float v) {
  // lowers to native global_atomic_add_f32 (no CAS loop) on gfx1250
  __hip_atomic_fetch_add(p, v, __ATOMIC_RELAXED, __HIP_MEMORY_SCOPE_AGENT);
}

// ---------------------------------------------------------------------------
__global__ void zero_f4(float* __restrict__ p, long n4) {
  long i = (long)blockIdx.x * blockDim.x + threadIdx.x;
  long stride = (long)gridDim.x * blockDim.x;
  float4 z = make_float4(0.f, 0.f, 0.f, 0.f);
  for (; i < n4; i += stride) ((float4*)p)[i] = z;
}

// agg1[dst] += feat[src] * mask1[src]   (in-dim = 1)
__global__ void scatter_edge1(const float* __restrict__ feat,
                              const float* __restrict__ mask1,
                              const int* __restrict__ src,
                              const int* __restrict__ dst,
                              float* __restrict__ agg1, int nE) {
  int e = blockIdx.x * blockDim.x + threadIdx.x;
  if (e >= nE) return;
  // keep the streaming index arrays ahead of the L2 atomics
  __builtin_prefetch(&src[e + 4096], 0, 1);
  __builtin_prefetch(&dst[e + 4096], 0, 1);
  int s = src[e];
  atomic_add_f32(&agg1[dst[e]], feat[s] * mask1[s]);
}

// h1[v][j] = relu(agg1[v] * W1[j] + b1[j]);  one thread = 4 features
__global__ void h1_kernel(const float* __restrict__ agg1,
                          const float* __restrict__ W1,
                          const float* __restrict__ b1,
                          float* __restrict__ h1, int nN) {
  int t = blockIdx.x * blockDim.x + threadIdx.x;
  int v = t >> 4;
  int jb = (t & 15) << 2;
  if (v >= nN) return;
  float a = agg1[v];
  float4 w = *(const float4*)(W1 + jb);
  float4 b = *(const float4*)(b1 + jb);
  float4 o;
  o.x = fmaxf(fmaf(a, w.x, b.x), 0.f);
  o.y = fmaxf(fmaf(a, w.y, b.y), 0.f);
  o.z = fmaxf(fmaf(a, w.z, b.z), 0.f);
  o.w = fmaxf(fmaf(a, w.w, b.w), 0.f);
  *(float4*)(h1 + (long)v * 64 + jb) = o;
}

// agg[dst, fb..fb+3] += xin[src, fb..fb+3];  16 threads per edge
__global__ void scatter_feat64(const float* __restrict__ xin,
                               const int* __restrict__ src,
                               const int* __restrict__ dst,
                               float* __restrict__ agg, long nWork) {
  long t = (long)blockIdx.x * blockDim.x + threadIdx.x;
  if (t >= nWork) return;
  long e = t >> 4;
  int fb = (int)(t & 15) << 2;
  __builtin_prefetch(&src[e + 1024], 0, 1);
  __builtin_prefetch(&dst[e + 1024], 0, 1);
  int s = src[e], d = dst[e];
  float4 v = *(const float4*)(xin + (long)s * 64 + fb);
  float* o = agg + (long)d * 64 + fb;
  atomic_add_f32(o + 0, v.x);
  atomic_add_f32(o + 1, v.y);
  atomic_add_f32(o + 2, v.z);
  atomic_add_f32(o + 3, v.w);
}

// ---------------------------------------------------------------------------
// WMMA fragment handling, per CDNA5 ISA 7.12.2 layouts (wave32).
//
// A (16x32 f16, M x K): lane<16 holds M=lane, K = {2vi,2vi+1} for vi<4 and
// {8+2vi, 9+2vi} for vi>=4; lanes 16..31 hold the same M with K offset +8.
__device__ __forceinline__ v16h load_a_frag(const float* __restrict__ act,
                                            int rowBase, int kk, int lane,
                                            int nN) {
  int m = rowBase + (lane & 15);
  m = m < nN ? m : nN - 1;               // branch-free clamp keeps EXEC full
  const float* arow = act + (long)m * 64;
  int koff = (lane < 16) ? 0 : 8;
  v16h a;
#pragma unroll
  for (int vi = 0; vi < 8; ++vi) {
    int k = kk + koff + ((vi < 4) ? (2 * vi) : (8 + 2 * vi));
    float2 f = *(const float2*)(arow + k);   // k even -> 8B aligned
    a[2 * vi]     = (_Float16)f.x;
    a[2 * vi + 1] = (_Float16)f.y;
  }
  return a;
}

// B fragments are pre-swizzled into LDS in fragment-major order:
// slot (frag = nt*2 + kblk, lane) holds the 16 halfs lane needs, so a B
// operand load is two ds_load_b128. Slot stride = 24 halfs (48 B): 16B
// aligned for b128 and bank-conflict-free across 16 lanes (12-bank stride).
#define BSLOT 24

// Fill sWf for a weight matrix W [64 x ldn] (f32, row-major), nFrags = (ldn/16)*2.
__device__ __forceinline__ void fill_b_frags(_Float16* __restrict__ sWf,
                                             const float* __restrict__ W,
                                             int ldn, int nFrags, int tid,
                                             int nthreads) {
  int total = nFrags * 32 * 16;
  for (int idx = tid; idx < total; idx += nthreads) {
    int frag = idx >> 9;            // /(32*16)
    int rem  = idx & 511;
    int lane = rem >> 4;
    int e    = rem & 15;            // e = 2*vi + half
    int nt   = frag >> 1;
    int kblk = frag & 1;
    int kgrp = (lane < 16) ? 0 : 8;
    int k    = kblk * 32 + kgrp + e + ((e < 8) ? 0 : 8);
    int n    = (nt << 4) + (lane & 15);
    sWf[(frag * 32 + lane) * BSLOT + e] = (_Float16)W[k * ldn + n];
  }
}

__device__ __forceinline__ v16h load_b_frag(const _Float16* __restrict__ sWf,
                                            int nt, int kblk, int lane) {
  int frag = nt * 2 + kblk;
  return *(const v16h*)(sWf + (frag * 32 + lane) * BSLOT);
}

// GEMM2 fused: x = relu(h1 + agg@W2 + b2) * mask2.   [N,64] x [64,64]
__global__ void __launch_bounds__(256)
gemm2_fused(const float* __restrict__ agg, const float* __restrict__ W2,
            const float* __restrict__ b2, const float* __restrict__ h1,
            const float* __restrict__ mask2, float* __restrict__ xout,
            int nN, int nTiles) {
  __shared__ __align__(16) _Float16 sWf[8 * 32 * BSLOT];   // 12 KB
  fill_b_frags(sWf, W2, 64, 8, threadIdx.x, blockDim.x);
  __syncthreads();

  int wave = threadIdx.x >> 5;
  int lane = threadIdx.x & 31;
  int tile = blockIdx.x * 8 + wave;
  if (tile >= nTiles) return;            // whole-wave uniform exit
  int rowBase = tile << 4;
  bool fullTile = (rowBase + 16) <= nN;  // wave-uniform

  v16h a0 = load_a_frag(agg, rowBase, 0, lane, nN);
  v16h a1 = load_a_frag(agg, rowBase, 32, lane, nN);

  int nlo  = lane & 15;
  int moff = (lane < 16) ? 0 : 8;        // D: M = r + 8*(lane>=16)

#pragma unroll
  for (int nt = 0; nt < 4; ++nt) {
    int nn = (nt << 4) + nlo;
    float bias = b2[nn];
    v8f c;
#pragma unroll
    for (int r = 0; r < 8; ++r) c[r] = bias;   // D = A*B + C, C preloaded w/ bias
    v16h bf0 = load_b_frag(sWf, nt, 0, lane);
    c = __builtin_amdgcn_wmma_f32_16x16x32_f16(false, a0, false, bf0,
                                               (short)0, c, false, false);
    v16h bf1 = load_b_frag(sWf, nt, 1, lane);
    c = __builtin_amdgcn_wmma_f32_16x16x32_f16(false, a1, false, bf1,
                                               (short)0, c, false, false);

    const float* h1p = h1 + (long)(rowBase + moff) * 64 + nn;
    const float* mkp = mask2 + (long)(rowBase + moff) * 64 + nn;
    float* xp = xout + (long)(rowBase + moff) * 64 + nn;
    if (fullTile) {
#pragma unroll
      for (int r = 0; r < 8; ++r) {
        float xv = h1p[(long)r * 64] + c[r];
        xp[(long)r * 64] = fmaxf(xv, 0.f) * mkp[(long)r * 64];
      }
    } else {
#pragma unroll
      for (int r = 0; r < 8; ++r) {
        int m = rowBase + moff + r;
        if (m < nN) {
          float xv = h1p[(long)r * 64] + c[r];
          xp[(long)r * 64] = fmaxf(xv, 0.f) * mkp[(long)r * 64];
        }
      }
    }
  }
}

// GEMM3: out = agg@W3 + b3.   [N,64] x [64,128]
__global__ void __launch_bounds__(256)
gemm3(const float* __restrict__ agg, const float* __restrict__ W3,
      const float* __restrict__ b3, float* __restrict__ out,
      int nN, int nTiles) {
  __shared__ __align__(16) _Float16 sWf[16 * 32 * BSLOT];  // 24 KB
  fill_b_frags(sWf, W3, 128, 16, threadIdx.x, blockDim.x);
  __syncthreads();

  int wave = threadIdx.x >> 5;
  int lane = threadIdx.x & 31;
  int tile = blockIdx.x * 8 + wave;
  if (tile >= nTiles) return;
  int rowBase = tile << 4;
  bool fullTile = (rowBase + 16) <= nN;

  v16h a0 = load_a_frag(agg, rowBase, 0, lane, nN);
  v16h a1 = load_a_frag(agg, rowBase, 32, lane, nN);

  int nlo  = lane & 15;
  int moff = (lane < 16) ? 0 : 8;

#pragma unroll
  for (int nt = 0; nt < 8; ++nt) {
    int nn = (nt << 4) + nlo;
    float bias = b3[nn];
    v8f c;
#pragma unroll
    for (int r = 0; r < 8; ++r) c[r] = bias;
    v16h bf0 = load_b_frag(sWf, nt, 0, lane);
    c = __builtin_amdgcn_wmma_f32_16x16x32_f16(false, a0, false, bf0,
                                               (short)0, c, false, false);
    v16h bf1 = load_b_frag(sWf, nt, 1, lane);
    c = __builtin_amdgcn_wmma_f32_16x16x32_f16(false, a1, false, bf1,
                                               (short)0, c, false, false);

    float* op = out + (long)(rowBase + moff) * 128 + nn;
    if (fullTile) {
#pragma unroll
      for (int r = 0; r < 8; ++r) op[(long)r * 128] = c[r];
    } else {
#pragma unroll
      for (int r = 0; r < 8; ++r) {
        int m = rowBase + moff + r;
        if (m < nN) op[(long)r * 128] = c[r];
      }
    }
  }
}

// ---------------------------------------------------------------------------
extern "C" void kernel_launch(void* const* d_in, const int* in_sizes, int n_in,
                              void* d_out, int out_size, void* d_ws,
                              size_t ws_size, hipStream_t stream) {
  const float* feat  = (const float*)d_in[0];
  const int*   src   = (const int*)d_in[1];
  const int*   dst   = (const int*)d_in[2];
  const float* W1    = (const float*)d_in[3];
  const float* b1    = (const float*)d_in[4];
  const float* W2    = (const float*)d_in[5];
  const float* b2    = (const float*)d_in[6];
  const float* W3    = (const float*)d_in[7];
  const float* b3    = (const float*)d_in[8];
  const float* mask1 = (const float*)d_in[9];
  const float* mask2 = (const float*)d_in[10];
  float* out = (float*)d_out;

  const int nN = in_sizes[0];   // features is [N,1]
  const int nE = in_sizes[1];

  // workspace layout (floats): h1 | agg (reused L2+L3) | x | agg1
  float* h1   = (float*)d_ws;
  float* agg  = h1  + (size_t)nN * 64;
  float* xbuf = agg + (size_t)nN * 64;
  float* agg1 = xbuf + (size_t)nN * 64;

  const int B = 256;
  int nTiles = (nN + 15) / 16;
  int gemmBlocks = (nTiles + 7) / 8;
  long scatterWork = (long)nE * 16;
  int scatterBlocks = (int)((scatterWork + B - 1) / B);

  // --- layer 1 ---
  {
    long n4 = (nN + 3) / 4;
    int zb = (int)((n4 + B - 1) / B);
    zero_f4<<<zb, B, 0, stream>>>(agg1, n4);
  }
  scatter_edge1<<<(nE + B - 1) / B, B, 0, stream>>>(feat, mask1, src, dst,
                                                    agg1, nE);
  h1_kernel<<<((long)nN * 16 + B - 1) / B, B, 0, stream>>>(agg1, W1, b1, h1,
                                                           nN);

  // --- layer 2 (fused residual + relu + dropout) ---
  {
    long n4 = (long)nN * 16;
    int zb = (int)((n4 + B - 1) / B);
    zero_f4<<<zb, B, 0, stream>>>(agg, n4);
  }
  scatter_feat64<<<scatterBlocks, B, 0, stream>>>(h1, src, dst, agg,
                                                  scatterWork);
  gemm2_fused<<<gemmBlocks, B, 0, stream>>>(agg, W2, b2, h1, mask2, xbuf, nN,
                                            nTiles);

  // --- layer 3 ---
  {
    long n4 = (long)nN * 16;
    int zb = (int)((n4 + B - 1) / B);
    zero_f4<<<zb, B, 0, stream>>>(agg, n4);
  }
  scatter_feat64<<<scatterBlocks, B, 0, stream>>>(xbuf, src, dst, agg,
                                                  scatterWork);
  gemm3<<<gemmBlocks, B, 0, stream>>>(agg, W3, b3, out, nN, nTiles);
}
